// MSHGAT_59167469469751
// MI455X (gfx1250) — matrix operand order repaired and checked by reading, not compiled, not run on hardware
//
#include <hip/hip_runtime.h>
#include <hip/hip_bf16.h>

typedef __attribute__((ext_vector_type(16))) _Float16 v16h;
typedef __attribute__((ext_vector_type(8)))  float    v8f;

#define Bq 32
#define Lq 128
#define Dq 72
#define DP 96                 // Dq padded to multiple of 32 (WMMA k-step)
#define NNODE 20000
#define NHD 512
#define ROWS (Bq*Lq)          // 4096
#define MASKW 640             // dwords per row of mask bitmap
#define NEGV (-4294967295.0f)

#define ZERO8 {0.f,0.f,0.f,0.f,0.f,0.f,0.f,0.f}

// ---------------- WMMA fragment helpers (CDNA5 16x16x32 f16 layouts) --------

__device__ __forceinline__ v8f wmma_f16(v16h a, v16h b, v8f c) {
  return __builtin_amdgcn_wmma_f32_16x16x32_f16(false, a, false, b, (short)0, c,
                                                false, false);
}

// A: 16x32 (MxK) f16 tile, row-major (K-contiguous), leading dim lda (elems).
// lane<16 : row=lane,   halves 0..7 = K 0..7,  halves 8..15 = K 16..23
// lane>=16: row=lane-16, halves 0..7 = K 8..15, halves 8..15 = K 24..31
__device__ __forceinline__ v16h load_a_frag(const _Float16* A, int lda) {
  int ln = threadIdx.x & 31;
  int m = ln & 15;
  int kb = (ln < 16) ? 0 : 8;
  union { v16h v; uint4 q[2]; } u;
  u.q[0] = *(const uint4*)(A + m * lda + kb);
  u.q[1] = *(const uint4*)(A + m * lda + kb + 16);
  return u.v;
}

// B (KxN) stored K-contiguous per column: Bt[n][k], leading dim ld (elems).
__device__ __forceinline__ v16h load_bt_frag(const _Float16* Bt, int ld) {
  int ln = threadIdx.x & 31;
  int n = ln & 15;
  int kb = (ln < 16) ? 0 : 8;
  union { v16h v; uint4 q[2]; } u;
  u.q[0] = *(const uint4*)(Bt + n * ld + kb);
  u.q[1] = *(const uint4*)(Bt + n * ld + kb + 16);
  return u.v;
}

// ---------------- embeddings (stride-DP activations, zero-padded) ------------

__global__ void embed_kernel(const int* __restrict__ tgt, const int* __restrict__ tsv,
                             const float* __restrict__ hidden, const float* __restrict__ memu,
                             const float* __restrict__ pos, float* __restrict__ diffE,
                             float* __restrict__ friE) {
  int g = blockIdx.x * blockDim.x + threadIdx.x;
  const int total = ROWS * DP;
  if (g >= total) return;
  int d = g % DP;
  int row = g / DP;
  int l = row % Lq, b = row / Lq;
  float fv, dv;
  if (d >= Dq) {                      // zero pad columns (keep WMMA NaN-free)
    fv = 0.f; dv = 0.f;
  } else if (d < 64) {
    int tok = tgt[b * 129 + l];
    fv = hidden[(size_t)tok * 64 + d];
    if (tok == 0) dv = 0.0f;
    else {
      int t = tsv[b * 129 + l];
      dv = (t == 0) ? hidden[(size_t)tok * 64 + d]
                    : memu[(size_t)(t - 1) * NNODE * 64 + (size_t)tok * 64 + d];
    }
  } else {
    float p = pos[l * 8 + (d - 64)];
    fv = p; dv = p;
  }
  diffE[(size_t)row * DP + d] = dv;
  friE [(size_t)row * DP + d] = fv;
}

// ---------------- generic WMMA GEMM: C = act(A@B + bias) + mask --------------
// A: MxK f32 (row stride lda, lda >= ceil32(K), pad cols finite),
// B: KxN f32 (dense), C: MxN f32 (row stride ldc; cols [N,ldc) zeroed).
// 64x64 tile per WG, 4 waves, each wave: 16-row strip x 4 col tiles.
// grid=(M/64, ceil(N/64)), block=128.

__global__ __launch_bounds__(128)
void gemm_wmma_kernel(const float* __restrict__ A, const float* __restrict__ Bm,
                      const float* __restrict__ bias, float* __restrict__ C,
                      int M, int N, int K, int lda, int ldc, int act,
                      const unsigned* __restrict__ rowmask) {
  __shared__ __align__(16) _Float16 sA [64 * 32];   // [m][k]
  __shared__ __align__(16) _Float16 sBt[64 * 32];   // [n][k]
  int m0 = blockIdx.x * 64;
  int n0 = blockIdx.y * 64;
  int w = threadIdx.x >> 5;
  v8f acc[4] = {ZERO8, ZERO8, ZERO8, ZERO8};

  for (int k0 = 0; k0 < K; k0 += 32) {
    // stage A (64x32): unconditional vector path (lda covers k0+c+16)
    {
      int r = threadIdx.x >> 1;
      int c = (threadIdx.x & 1) * 16;
      const float* src = A + (size_t)(m0 + r) * lda + k0 + c;
      union { _Float16 h[16]; uint4 q[2]; } t;
#pragma unroll
      for (int i = 0; i < 16; ++i) t.h[i] = (_Float16)src[i];
      uint4* dst = (uint4*)(sA + r * 32 + c);
      dst[0] = t.q[0];
      dst[1] = t.q[1];
    }
    // stage B transposed (coalesced global read, scatter to [n][k])
    for (int idx = threadIdx.x; idx < 32 * 64; idx += 128) {
      int r = idx >> 6, cc = idx & 63;           // r = k local, cc = n local
      int gk = k0 + r, gn = n0 + cc;
      float v = (gk < K && gn < N) ? Bm[(size_t)gk * N + gn] : 0.0f;
      sBt[cc * 32 + r] = (_Float16)v;
    }
    if (k0 + 32 < K)   // hint next B panel toward cache (global_prefetch)
      __builtin_prefetch(&Bm[(size_t)(k0 + 32) * N + n0 + (threadIdx.x & 63)], 0, 1);
    __syncthreads();
    v16h a = load_a_frag(sA + (w * 16) * 32, 32);
#pragma unroll
    for (int ct = 0; ct < 4; ++ct) {
      v16h b = load_bt_frag(sBt + (ct * 16) * 32, 32);
      acc[ct] = wmma_f16(a, b, acc[ct]);
    }
    __syncthreads();
  }

  int ln = threadIdx.x & 31;
  int nloc = ln & 15, rb = (ln < 16) ? 0 : 8;
#pragma unroll
  for (int ct = 0; ct < 4; ++ct) {
    int n = n0 + ct * 16 + nloc;
    if (n < N) {
      float bv = bias ? bias[n] : 0.0f;
#pragma unroll
      for (int r = 0; r < 8; ++r) {
        int m = m0 + w * 16 + rb + r;
        float v = acc[ct][r] + bv;
        if (act == 1 && v < 0.f) v = 0.f;
        if (rowmask) {
          unsigned mw = rowmask[(size_t)m * MASKW + (n >> 5)];
          if ((mw >> (n & 31)) & 1u) v += -1000.0f;
        }
        C[(size_t)m * ldc + n] = v;
      }
    } else if (n < ldc) {            // zero the pad columns of padded outputs
#pragma unroll
      for (int r = 0; r < 8; ++r)
        C[(size_t)(m0 + w * 16 + rb + r) * ldc + n] = 0.0f;
    }
  }
}

// ---------------- attention: scores, softmax, att@V --------------------------

__global__ __launch_bounds__(256)
void scores_kernel(const float* __restrict__ Q, const float* __restrict__ Km,
                   const int* __restrict__ tgt, float* __restrict__ S) {
  __shared__ __align__(16) _Float16 sQ[Lq * 64];   // [l][d]
  __shared__ __align__(16) _Float16 sK[Lq * 64];   // [l][d] (acts as Bt: k^T)
  int bh = blockIdx.x, b = bh >> 3, h = bh & 7;
  {
    int l = threadIdx.x >> 1, part = (threadIdx.x & 1) * 32;
    const float* qsrc = Q  + (size_t)(b * Lq + l) * NHD + h * 64 + part;
    const float* ksrc = Km + (size_t)(b * Lq + l) * NHD + h * 64 + part;
    union { _Float16 h[32]; uint4 q[4]; } tq, tk;
#pragma unroll
    for (int i = 0; i < 32; ++i) { tq.h[i] = (_Float16)qsrc[i]; tk.h[i] = (_Float16)ksrc[i]; }
    uint4* dq = (uint4*)(sQ + l * 64 + part);
    uint4* dk = (uint4*)(sK + l * 64 + part);
#pragma unroll
    for (int j = 0; j < 4; ++j) { dq[j] = tq.q[j]; dk[j] = tk.q[j]; }
  }
  __syncthreads();
  int w = threadIdx.x >> 5, ln = threadIdx.x & 31;
  int nloc = ln & 15, rb = (ln < 16) ? 0 : 8;
  const float scale = 1.0f / (8.0f + 1e-6f);
  v8f acc[8] = {ZERO8, ZERO8, ZERO8, ZERO8, ZERO8, ZERO8, ZERO8, ZERO8};
#pragma unroll
  for (int k0 = 0; k0 < 64; k0 += 32) {
    v16h a = load_a_frag(sQ + (w * 16) * 64 + k0, 64);
#pragma unroll
    for (int ct = 0; ct < 8; ++ct) {
      v16h bt = load_bt_frag(sK + (ct * 16) * 64 + k0, 64);
      acc[ct] = wmma_f16(a, bt, acc[ct]);
    }
  }
  float* dst = S + (size_t)bh * Lq * Lq;
#pragma unroll
  for (int ct = 0; ct < 8; ++ct) {
    int kcol = ct * 16 + nloc;
#pragma unroll
    for (int r = 0; r < 8; ++r) {
      int qrow = w * 16 + rb + r;
      float v = acc[ct][r] * scale;
      if (kcol > qrow || tgt[b * 129 + qrow] == 0) v = NEGV;
      dst[(size_t)qrow * Lq + kcol] = v;
    }
  }
}

__global__ __launch_bounds__(256)
void softmax_kernel(const float* __restrict__ S, _Float16* __restrict__ P) {
  int wave = threadIdx.x >> 5, ln = threadIdx.x & 31;
  size_t row = (size_t)blockIdx.x * 8 + wave;   // 32768 rows
  const float* src = S + row * Lq;
  float x[4], m = -3.4e38f;
#pragma unroll
  for (int j = 0; j < 4; ++j) { x[j] = src[ln + 32 * j]; m = fmaxf(m, x[j]); }
#pragma unroll
  for (int off = 16; off > 0; off >>= 1) m = fmaxf(m, __shfl_xor(m, off, 32));
  float e[4], s = 0.f;
#pragma unroll
  for (int j = 0; j < 4; ++j) { e[j] = __expf(x[j] - m); s += e[j]; }
#pragma unroll
  for (int off = 16; off > 0; off >>= 1) s += __shfl_xor(s, off, 32);
  float inv = 1.0f / s;
  _Float16* dst = P + row * Lq;
#pragma unroll
  for (int j = 0; j < 4; ++j) dst[ln + 32 * j] = (_Float16)(e[j] * inv);
}

__global__ __launch_bounds__(256)
void av_kernel(const _Float16* __restrict__ P, const float* __restrict__ V,
               float* __restrict__ O) {
  __shared__ __align__(16) _Float16 sP [Lq * Lq];  // [q][k]
  __shared__ __align__(16) _Float16 sVt[64 * Lq];  // [d][k] transposed
  int bh = blockIdx.x, b = bh >> 3, h = bh & 7;
  {
    const uint4* psrc = (const uint4*)(P + (size_t)bh * Lq * Lq);
    uint4* pdst = (uint4*)sP;
    for (int j = threadIdx.x; j < Lq * Lq / 8; j += 256) pdst[j] = psrc[j];
  }
  for (int idx = threadIdx.x; idx < Lq * 64; idx += 256) {
    int l = idx >> 6, d = idx & 63;
    sVt[d * Lq + l] = (_Float16)V[(size_t)(b * Lq + l) * NHD + h * 64 + d];
  }
  __syncthreads();
  int w = threadIdx.x >> 5, ln = threadIdx.x & 31;
  int nloc = ln & 15, rb = (ln < 16) ? 0 : 8;
  v8f acc[4] = {ZERO8, ZERO8, ZERO8, ZERO8};
#pragma unroll
  for (int k0 = 0; k0 < Lq; k0 += 32) {
    v16h a = load_a_frag(sP + (w * 16) * Lq + k0, Lq);
#pragma unroll
    for (int ct = 0; ct < 4; ++ct) {
      v16h bb = load_bt_frag(sVt + (ct * 16) * Lq + k0, Lq);
      acc[ct] = wmma_f16(a, bb, acc[ct]);
    }
  }
#pragma unroll
  for (int ct = 0; ct < 4; ++ct) {
#pragma unroll
    for (int r = 0; r < 8; ++r) {
      int qrow = w * 16 + rb + r;
      O[(size_t)(b * Lq + qrow) * NHD + h * 64 + ct * 16 + nloc] = acc[ct][r];
    }
  }
}

// ---------------- residual + LayerNorm (stride-DP in/out, zero pads) ---------

__global__ void add_ln_kernel(const float* __restrict__ A, const float* __restrict__ Bd,
                              const float* __restrict__ g, const float* __restrict__ bb,
                              float* __restrict__ out) {
  int row = blockIdx.x * blockDim.x + threadIdx.x;
  if (row >= ROWS) return;
  float x[Dq], s = 0.f;
#pragma unroll
  for (int d = 0; d < Dq; ++d) { x[d] = A[(size_t)row * DP + d] + Bd[(size_t)row * DP + d]; s += x[d]; }
  float m = s / (float)Dq, v = 0.f;
#pragma unroll
  for (int d = 0; d < Dq; ++d) { float t = x[d] - m; v += t * t; }
  v /= (float)Dq;
  float inv = rsqrtf(v + 1e-5f);
#pragma unroll
  for (int d = 0; d < Dq; ++d)
    out[(size_t)row * DP + d] = (x[d] - m) * inv * g[d] + bb[d];
#pragma unroll
  for (int d = Dq; d < DP; ++d)
    out[(size_t)row * DP + d] = 0.f;
}

// ---------------- two-way gated fusion (stride-DP, zero pads) ----------------

__global__ void fusion_kernel(const float* __restrict__ e0, const float* __restrict__ e1,
                              const float* __restrict__ w1, const float* __restrict__ b1,
                              const float* __restrict__ w2, const float* __restrict__ b2,
                              float* __restrict__ out) {
  __shared__ float red[128];
  __shared__ float ush[2];
  int row = blockIdx.x, tid = threadIdx.x;
  for (int s = 0; s < 2; ++s) {
    const float* e = (s ? e1 : e0) + (size_t)row * DP;
    float contrib = 0.f;
    if (tid < Dq) {
      float acc = b1[tid];
      for (int k = 0; k < Dq; ++k) acc += e[k] * w1[k * Dq + tid];
      contrib = tanhf(acc) * w2[tid];
    }
    red[tid] = contrib;
    __syncthreads();
    for (int st = 64; st > 0; st >>= 1) {
      if (tid < st) red[tid] += red[tid + st];
      __syncthreads();
    }
    if (tid == 0) ush[s] = red[0] + b2[0];
    __syncthreads();
  }
  float u0 = ush[0], u1 = ush[1];
  float mx = fmaxf(u0, u1);
  float a0 = __expf(u0 - mx), a1 = __expf(u1 - mx);
  float w0 = a0 / (a0 + a1), wq = a1 / (a0 + a1);
  if (tid < Dq)
    out[(size_t)row * DP + tid] =
        w0 * e0[(size_t)row * DP + tid] + wq * e1[(size_t)row * DP + tid];
  else if (tid < DP)
    out[(size_t)row * DP + tid] = 0.f;
}

// ---------------- prefix-token mask bitmap -----------------------------------

__global__ void mark_kernel(const int* __restrict__ tgt, unsigned* __restrict__ bitmap) {
  int bl = blockIdx.x;           // row index 0..4095
  int l = bl % Lq, b = bl / Lq;
  int j = threadIdx.x;           // 0..127
  unsigned* rowm = bitmap + (size_t)bl * MASKW;
  if (j == 0) atomicOr(&rowm[0], 1u);          // node 0 always masked
  if (j <= l) {
    int node = tgt[b * 129 + j];
    atomicOr(&rowm[node >> 5], 1u << (node & 31));
  }
}

// ---------------- host orchestration -----------------------------------------

extern "C" void kernel_launch(void* const* d_in, const int* in_sizes, int n_in,
                              void* d_out, int out_size, void* d_ws, size_t ws_size,
                              hipStream_t stream) {
  const int*   tgt   = (const int*)d_in[0];
  const int*   tsv   = (const int*)d_in[1];
  const float* hidden  = (const float*)d_in[4];
  const float* memu    = (const float*)d_in[5];
  const float* pos     = (const float*)d_in[7];
  const float* aqkv    = (const float*)d_in[8];
  const float* ao      = (const float*)d_in[9];
  const float* aob     = (const float*)d_in[10];
  const float* fw      = (const float*)d_in[11];
  const float* fb      = (const float*)d_in[12];
  const float* lng     = (const float*)d_in[13];
  const float* lnb     = (const float*)d_in[14];
  const float* fw1     = (const float*)d_in[15];
  const float* fb1     = (const float*)d_in[16];
  const float* fw2     = (const float*)d_in[17];
  const float* fb2     = (const float*)d_in[18];
  const float* lin2w   = (const float*)d_in[19];
  const float* lin2b   = (const float*)d_in[20];
  float* outp = (float*)d_out;

  size_t off = 0;
  auto alloc = [&](size_t bytes) -> void* {
    off = (off + 255) & ~(size_t)255;
    void* p = (char*)d_ws + off;
    off += bytes;
    return p;
  };
  float*     diffE  = (float*)alloc((size_t)ROWS * DP * 4);
  float*     friE   = (float*)alloc((size_t)ROWS * DP * 4);
  float*     bufQ   = (float*)alloc((size_t)ROWS * NHD * 4);
  float*     bufK   = (float*)alloc((size_t)ROWS * NHD * 4);
  float*     bufV   = (float*)alloc((size_t)ROWS * NHD * 4);
  float*     bufO   = (float*)alloc((size_t)ROWS * NHD * 4);
  float*     bufS   = (float*)alloc((size_t)256 * Lq * Lq * 4);
  _Float16*  bufP   = (_Float16*)alloc((size_t)256 * Lq * Lq * 2);
  float*     bufT1  = (float*)alloc((size_t)ROWS * DP * 4);
  float*     bufT2  = (float*)alloc((size_t)ROWS * DP * 4);
  float*     bufX   = (float*)alloc((size_t)ROWS * DP * 4);
  float*     diffA  = (float*)alloc((size_t)ROWS * DP * 4);
  float*     friA   = (float*)alloc((size_t)ROWS * DP * 4);
  float*     attOut = (float*)alloc((size_t)ROWS * DP * 4);
  unsigned*  bitmap = (unsigned*)alloc((size_t)ROWS * MASKW * 4);

  // 1) embeddings (reference T-loop collapses to a gather; cas_emb is unused)
  {
    int total = ROWS * DP;
    embed_kernel<<<(total + 255) / 256, 256, 0, stream>>>(tgt, tsv, hidden, memu,
                                                          pos, diffE, friE);
  }

  // 2) two transformer blocks
  for (int i = 0; i < 2; ++i) {
    const float* E   = (i == 0) ? diffE : friE;
    float*       Out = (i == 0) ? diffA : friA;
    const float* Wq = aqkv + (size_t)i * 3 * Dq * NHD;
    const float* Wk = Wq + (size_t)Dq * NHD;
    const float* Wv = Wk + (size_t)Dq * NHD;

    dim3 gQKV(ROWS / 64, (NHD + 63) / 64);
    gemm_wmma_kernel<<<gQKV, 128, 0, stream>>>(E, Wq, nullptr, bufQ, ROWS, NHD, Dq, DP, NHD, 0, nullptr);
    gemm_wmma_kernel<<<gQKV, 128, 0, stream>>>(E, Wk, nullptr, bufK, ROWS, NHD, Dq, DP, NHD, 0, nullptr);
    gemm_wmma_kernel<<<gQKV, 128, 0, stream>>>(E, Wv, nullptr, bufV, ROWS, NHD, Dq, DP, NHD, 0, nullptr);

    scores_kernel<<<256, 256, 0, stream>>>(bufQ, bufK, tgt, bufS);
    softmax_kernel<<<4096, 256, 0, stream>>>(bufS, bufP);
    av_kernel<<<256, 256, 0, stream>>>(bufP, bufV, bufO);

    dim3 gPro(ROWS / 64, (Dq + 63) / 64);
    gemm_wmma_kernel<<<gPro, 128, 0, stream>>>(bufO, ao + (size_t)i * NHD * Dq,
                                               aob + (size_t)i * Dq, bufT1,
                                               ROWS, Dq, NHD, NHD, DP, 0, nullptr);
    add_ln_kernel<<<ROWS / 128, 128, 0, stream>>>(E, bufT1, lng + i * Dq, lnb + i * Dq, bufX);

    gemm_wmma_kernel<<<gPro, 128, 0, stream>>>(bufX, fw + (size_t)(i * 2 + 0) * Dq * Dq,
                                               fb + (size_t)(i * 2 + 0) * Dq, bufT1,
                                               ROWS, Dq, Dq, DP, DP, 1, nullptr);
    gemm_wmma_kernel<<<gPro, 128, 0, stream>>>(bufT1, fw + (size_t)(i * 2 + 1) * Dq * Dq,
                                               fb + (size_t)(i * 2 + 1) * Dq, bufT2,
                                               ROWS, Dq, Dq, DP, DP, 0, nullptr);
    add_ln_kernel<<<ROWS / 128, 128, 0, stream>>>(bufX, bufT2, lng + i * Dq, lnb + i * Dq, Out);
  }

  // 3) fusion
  fusion_kernel<<<ROWS, 128, 0, stream>>>(diffA, friA, fw1, fb1, fw2, fb2, attOut);

  // 4) prefix mask bitmap
  hipMemsetAsync(bitmap, 0, (size_t)ROWS * MASKW * 4, stream);
  mark_kernel<<<ROWS, 128, 0, stream>>>(tgt, bitmap);

  // 5) final logits GEMM (M=4096,K=72,N=20000) with bias + (-1000) mask fused;
  //    this kernel's 327MB output write is the roofline limit (~14us @ 23.3TB/s)
  dim3 gFin(ROWS / 64, (NNODE + 63) / 64);
  gemm_wmma_kernel<<<gFin, 128, 0, stream>>>(attOut, lin2w, lin2b, outp,
                                             ROWS, NNODE, Dq, DP, NNODE, 0, bitmap);
  (void)in_sizes; (void)n_in; (void)out_size; (void)ws_size;
}
